// PointNetEncoder_37598143709460
// MI455X (gfx1250) — matrix operand order repaired and checked by reading, not compiled, not run on hardware
//
#include <hip/hip_runtime.h>
#include <hip/hip_bf16.h>
#include <cstdint>

typedef __attribute__((ext_vector_type(16))) _Float16 v16h;
typedef __attribute__((ext_vector_type(8)))  _Float16 v8h;
typedef __attribute__((ext_vector_type(8)))  float    v8f;

union V16H { v16h v; v8h q[2]; _Float16 h[16]; };

// ---------------------------------------------------------------------------
// WMMA GEMM:  Y[M,N] = Xh[M,Kp] * Wh[N,Kp]^T + bias[N]
// Xh/Wh f16, Kp multiple of 32; M multiple of 16; Nn multiple of 64.
// One wave32 computes a 16x64 strip (4 N-tiles) so the A panel is loaded
// once per strip: per K-step = 2 A b128 loads + 8 B b128 loads + 4 WMMAs.
// A layout (16x32 f16): lane l (row=l&15, half=l>>4), elem e ->
//   k = half*16 + e
// B layout (32x16 f16): lane l (col=l&15, half=l>>4), elem e -> k = half*16+e
// C/D f32: lane l col=l&15; vgpr v row = v + (l>>4)*8
// ---------------------------------------------------------------------------
__global__ void gemm_f16_wmma(const _Float16* __restrict__ Xh,
                              const _Float16* __restrict__ Wh,
                              const float* __restrict__ bias,
                              float* __restrict__ Y,
                              int M, int Kp, int Nn) {
  const int lane    = threadIdx.x & 31;
  const int wave    = threadIdx.x >> 5;
  const int stripsN = Nn >> 6;                       // 64-wide strips
  const int strip   = blockIdx.x * (blockDim.x >> 5) + wave;
  if (strip >= (M >> 4) * stripsN) return;           // wave-uniform
  const int tm    = (strip / stripsN) << 4;
  const int tn    = (strip % stripsN) << 6;
  const int l15   = lane & 15;
  const int khalf = lane >> 4;

  const _Float16* ap = Xh + (size_t)(tm + l15) * Kp + (khalf << 4);
  const _Float16* bp = Wh + (size_t)(tn + l15) * Kp + (khalf << 4);
  const size_t bstr  = (size_t)16 * Kp;              // next 16-col W tile

  v8f acc0 = {}, acc1 = {}, acc2 = {}, acc3 = {};
  for (int k0 = 0; k0 < Kp; k0 += 32) {
    V16H a, b0, b1, b2, b3;
    a.q[0]  = *reinterpret_cast<const v8h*>(ap + k0);
    a.q[1]  = *reinterpret_cast<const v8h*>(ap + k0 + 8);
    b0.q[0] = *reinterpret_cast<const v8h*>(bp + k0);
    b0.q[1] = *reinterpret_cast<const v8h*>(bp + k0 + 8);
    b1.q[0] = *reinterpret_cast<const v8h*>(bp + bstr + k0);
    b1.q[1] = *reinterpret_cast<const v8h*>(bp + bstr + k0 + 8);
    b2.q[0] = *reinterpret_cast<const v8h*>(bp + 2 * bstr + k0);
    b2.q[1] = *reinterpret_cast<const v8h*>(bp + 2 * bstr + k0 + 8);
    b3.q[0] = *reinterpret_cast<const v8h*>(bp + 3 * bstr + k0);
    b3.q[1] = *reinterpret_cast<const v8h*>(bp + 3 * bstr + k0 + 8);
    acc0 = __builtin_amdgcn_wmma_f32_16x16x32_f16(false, a.v, false, b0.v,
                                                  (short)0, acc0, false, false);
    acc1 = __builtin_amdgcn_wmma_f32_16x16x32_f16(false, a.v, false, b1.v,
                                                  (short)0, acc1, false, false);
    acc2 = __builtin_amdgcn_wmma_f32_16x16x32_f16(false, a.v, false, b2.v,
                                                  (short)0, acc2, false, false);
    acc3 = __builtin_amdgcn_wmma_f32_16x16x32_f16(false, a.v, false, b3.v,
                                                  (short)0, acc3, false, false);
  }

  float* yp = Y + (size_t)(tm + (khalf << 3)) * Nn + tn + l15;
  const float bv0 = bias[tn + l15];
  const float bv1 = bias[tn + 16 + l15];
  const float bv2 = bias[tn + 32 + l15];
  const float bv3 = bias[tn + 48 + l15];
#pragma unroll
  for (int v = 0; v < 8; ++v) {
    float* r = yp + (size_t)v * Nn;
    r[0]  = acc0[v] + bv0;
    r[16] = acc1[v] + bv1;
    r[32] = acc2[v] + bv2;
    r[48] = acc3[v] + bv3;
  }
}

// ---------------------------------------------------------------------------
// Farthest point sampling: one block per batch; dist[] in LDS; wave32 argmax.
// ---------------------------------------------------------------------------
__global__ void fps_kernel(const float* __restrict__ xyz, int* __restrict__ fidx,
                           int N, int Snp) {
  const int b = blockIdx.x;
  const float* p = xyz + (size_t)b * N * 3;
  __shared__ float dist[4096];
  __shared__ float swv[8];
  __shared__ int   swi[8];
  __shared__ int   sfar;
  for (int n = threadIdx.x; n < N; n += blockDim.x) dist[n] = 1e10f;
  if (threadIdx.x == 0) sfar = 0;
  __syncthreads();

  for (int j = 0; j < Snp; ++j) {
    const int far = sfar;
    if (threadIdx.x == 0) fidx[b * Snp + j] = far;
    const float cx = p[far * 3 + 0], cy = p[far * 3 + 1], cz = p[far * 3 + 2];

    float bestv = -1.0f; int besti = 0x7fffffff;
    for (int n = threadIdx.x; n < N; n += blockDim.x) {
      const float dx = p[n * 3 + 0] - cx;
      const float dy = p[n * 3 + 1] - cy;
      const float dz = p[n * 3 + 2] - cz;
      float d = dx * dx + dy * dy + dz * dz;
      d = fminf(dist[n], d);
      dist[n] = d;
      if (d > bestv || (d == bestv && n < besti)) { bestv = d; besti = n; }
    }
    for (int off = 16; off > 0; off >>= 1) {
      const float ov = __shfl_xor(bestv, off);
      const int   oi = __shfl_xor(besti, off);
      if (ov > bestv || (ov == bestv && oi < besti)) { bestv = ov; besti = oi; }
    }
    const int wv = threadIdx.x >> 5;
    if ((threadIdx.x & 31) == 0) { swv[wv] = bestv; swi[wv] = besti; }
    __syncthreads();
    if (threadIdx.x == 0) {
      float bv = swv[0]; int bi = swi[0];
      const int nw = blockDim.x >> 5;
      for (int w = 1; w < nw; ++w)
        if (swv[w] > bv || (swv[w] == bv && swi[w] < bi)) { bv = swv[w]; bi = swi[w]; }
      sfar = bi;
    }
    __syncthreads();
  }
}

__global__ void gather_nxyz_kernel(const float* __restrict__ xyz,
                                   const int* __restrict__ fidx,
                                   float* __restrict__ nxyz,
                                   int B, int N, int S) {
  const int t = blockIdx.x * blockDim.x + threadIdx.x;
  if (t >= B * S * 3) return;
  const int c = t % 3, bs = t / 3;
  const int b = bs / S;
  nxyz[t] = xyz[((size_t)b * N + fidx[bs]) * 3 + c];
}

__global__ void ballq_kernel(const float* __restrict__ xyz,
                             const float* __restrict__ nxyz,
                             int* __restrict__ gidx,
                             int B, int N, int S, int K, float r2) {
  const int t = blockIdx.x * blockDim.x + threadIdx.x;
  if (t >= B * S) return;
  const int b = t / S;
  const float* p = xyz + (size_t)b * N * 3;
  const float qx = nxyz[t * 3 + 0], qy = nxyz[t * 3 + 1], qz = nxyz[t * 3 + 2];
  int* out = gidx + (size_t)t * K;
  int cnt = 0, first = 0;
  for (int n = 0; n < N && cnt < K; ++n) {
    const float dx = p[n * 3 + 0] - qx;
    const float dy = p[n * 3 + 1] - qy;
    const float dz = p[n * 3 + 2] - qz;
    if (dx * dx + dy * dy + dz * dz <= r2) {
      if (cnt == 0) first = n;
      out[cnt++] = n;
    }
  }
  for (; cnt < K; ++cnt) out[cnt] = first;
}

// Gather + center + f16-pack (Kp=32): one thread per group point, 4x 16B stores.
__global__ void group_pack_kernel(const float* __restrict__ xyz,
                                  const float* __restrict__ nxyz,
                                  const int* __restrict__ gidx,
                                  _Float16* __restrict__ Xh,
                                  int B, int N, int S, int K) {
  const int t = blockIdx.x * blockDim.x + threadIdx.x;
  if (t >= B * S * K) return;
  const int bs = t / K;
  const int b = bs / S;
  const int idx = gidx[t];
  v8h z = {};
  v8h r0 = z;
  r0.s0 = (_Float16)(xyz[((size_t)b * N + idx) * 3 + 0] - nxyz[(size_t)bs * 3 + 0]);
  r0.s1 = (_Float16)(xyz[((size_t)b * N + idx) * 3 + 1] - nxyz[(size_t)bs * 3 + 1]);
  r0.s2 = (_Float16)(xyz[((size_t)b * N + idx) * 3 + 2] - nxyz[(size_t)bs * 3 + 2]);
  v8h* dst = reinterpret_cast<v8h*>(Xh + (size_t)t * 32);
  dst[0] = r0; dst[1] = z; dst[2] = z; dst[3] = z;
}

// Weight f16 pack with K padding.
__global__ void pack_w_kernel(const float* __restrict__ W, _Float16* __restrict__ Wh,
                              int Nn, int Kd, int Kp) {
  const int t = blockIdx.x * blockDim.x + threadIdx.x;
  if (t >= Nn * Kp) return;
  const int n = t / Kp, k = t % Kp;
  Wh[t] = (k < Kd) ? (_Float16)W[(size_t)n * Kd + k] : (_Float16)0.0f;
}

// ---------------------------------------------------------------------------
// BN stats: one block per channel; deterministic tree reduction.
// a[o] = rsqrt(var+eps)*g[o], c[o] = beta[o] - mean*a[o]
// ---------------------------------------------------------------------------
__global__ void stats_kernel(const float* __restrict__ Y, int M, int C,
                             const float* __restrict__ g,
                             const float* __restrict__ beta,
                             float* __restrict__ A, float* __restrict__ Cc) {
  const int o = blockIdx.x;
  __shared__ float ss[256], sq[256];
  float s = 0.0f, q = 0.0f;
  for (int m = threadIdx.x; m < M; m += blockDim.x) {
    const float v = Y[(size_t)m * C + o];
    s += v; q += v * v;
  }
  ss[threadIdx.x] = s; sq[threadIdx.x] = q;
  __syncthreads();
  for (int off = 128; off > 0; off >>= 1) {
    if ((int)threadIdx.x < off) {
      ss[threadIdx.x] += ss[threadIdx.x + off];
      sq[threadIdx.x] += sq[threadIdx.x + off];
    }
    __syncthreads();
  }
  if (threadIdx.x == 0) {
    const float mean = ss[0] / (float)M;
    const float var  = sq[0] / (float)M - mean * mean;
    const float av   = rsqrtf(var + 1e-5f) * g[o];
    A[o]  = av;
    Cc[o] = beta[o] - mean * av;
  }
}

// Fused BN + ReLU + f16 pack (writes next layer's GEMM input, K-padded).
__global__ void normrelu_pack_kernel(const float* __restrict__ Y,
                                     const float* __restrict__ A,
                                     const float* __restrict__ Cc,
                                     _Float16* __restrict__ Xh,
                                     int M, int C, int Kp) {
  const long long total = (long long)M * Kp;
  const long long i = (long long)blockIdx.x * blockDim.x + threadIdx.x;
  if (i >= total) return;
  const int j = (int)(i % Kp);
  const long long m = i / Kp;
  float v = 0.0f;
  if (j < C) v = fmaxf(0.0f, Y[m * C + j] * A[j] + Cc[j]);
  Xh[i] = (_Float16)v;
}

// Fused BN + ReLU + max-pool over the K grouping axis.
__global__ void normrelu_maxpool_kernel(const float* __restrict__ Y,
                                        const float* __restrict__ A,
                                        const float* __restrict__ Cc,
                                        float* __restrict__ out,
                                        int Rows, int K, int C) {
  const int t = blockIdx.x * blockDim.x + threadIdx.x;
  if (t >= Rows * C) return;
  const int r = t / C, c = t % C;
  const float a = A[c], cc = Cc[c];
  float m = -1e30f;
  const float* yp = Y + (size_t)r * K * C + c;
  for (int k = 0; k < K; ++k)
    m = fmaxf(m, fmaxf(0.0f, yp[(size_t)k * C] * a + cc));
  out[(size_t)r * C + c] = m;
}

// Concat l1_xyz(3) + l1_pts(128) -> f16 padded to Kp=160.
__global__ void concat_pack_kernel(const float* __restrict__ nxyz,
                                   const float* __restrict__ pts,
                                   _Float16* __restrict__ Xh,
                                   int BS, int Cp /*128*/, int Kp /*160*/) {
  const int t = blockIdx.x * blockDim.x + threadIdx.x;
  if (t >= BS * Kp) return;
  const int row = t / Kp, j = t % Kp;
  float v = 0.0f;
  if (j < 3)            v = nxyz[(size_t)row * 3 + j];
  else if (j < 3 + Cp)  v = pts[(size_t)row * Cp + (j - 3)];
  Xh[t] = (_Float16)v;
}

// ---------------------------------------------------------------------------
extern "C" void kernel_launch(void* const* d_in, const int* in_sizes, int n_in,
                              void* d_out, int out_size, void* d_ws, size_t ws_size,
                              hipStream_t stream) {
  (void)in_sizes; (void)n_in; (void)out_size; (void)ws_size;
  const int B = 32, N = 4096, S = 512, K = 32;
  const float radius = 0.2f;
  const int BS = B * S;           // 16384
  const int M1 = BS * K;          // 524288

  const float* xyz = (const float*)d_in[0];
  const float *w[6], *bs_[6], *gg[6], *bb[6];
  for (int L = 0; L < 3; ++L) {
    w[L]      = (const float*)d_in[1 + L * 4 + 0];
    bs_[L]    = (const float*)d_in[1 + L * 4 + 1];
    gg[L]     = (const float*)d_in[1 + L * 4 + 2];
    bb[L]     = (const float*)d_in[1 + L * 4 + 3];
    w[3 + L]  = (const float*)d_in[13 + L * 4 + 0];
    bs_[3 + L]= (const float*)d_in[13 + L * 4 + 1];
    gg[3 + L] = (const float*)d_in[13 + L * 4 + 2];
    bb[3 + L] = (const float*)d_in[13 + L * 4 + 3];
  }
  // layer dims: {Kd, Kp, Nn}
  const int Kd[6] = {3, 64, 64, 131, 128, 128};
  const int Kp[6] = {32, 64, 64, 160, 128, 128};
  const int Nn[6] = {64, 64, 128, 128, 128, 256};

  size_t off = 0;
  auto carve = [&](size_t bytes) -> void* {
    void* p = (char*)d_ws + off;
    off += (bytes + 255) & ~(size_t)255;
    return p;
  };
  int*       fidx  = (int*)      carve((size_t)BS * sizeof(int));
  int*       gidx  = (int*)      carve((size_t)BS * K * sizeof(int));
  float*     nxyz  = (float*)    carve((size_t)BS * 3 * sizeof(float));
  _Float16*  XhA   = (_Float16*) carve((size_t)M1 * 64 * sizeof(_Float16));
  _Float16*  XhB   = (_Float16*) carve((size_t)M1 * 64 * sizeof(_Float16));
  float*     Ybuf  = (float*)    carve((size_t)M1 * 128 * sizeof(float));
  float*     acoef = (float*)    carve(256 * sizeof(float));
  float*     ccoef = (float*)    carve(256 * sizeof(float));
  float*     l1pts = (float*)    carve((size_t)BS * 128 * sizeof(float));
  _Float16*  Wh[6];
  for (int L = 0; L < 6; ++L)
    Wh[L] = (_Float16*)carve((size_t)Nn[L] * Kp[L] * sizeof(_Float16));

  auto cdiv = [](long long a, long long b) { return (int)((a + b - 1) / b); };

  // Pack all weights to padded f16.
  for (int L = 0; L < 6; ++L)
    pack_w_kernel<<<cdiv((long long)Nn[L] * Kp[L], 256), 256, 0, stream>>>(
        w[L], Wh[L], Nn[L], Kd[L], Kp[L]);

  auto gemm = [&](const _Float16* X, int L, float* Y, int M) {
    const long long strips = (long long)(M >> 4) * (Nn[L] >> 6);
    gemm_f16_wmma<<<cdiv(strips, 8), 256, 0, stream>>>(X, Wh[L], bs_[L], Y,
                                                       M, Kp[L], Nn[L]);
  };
  auto bnstats = [&](float* Y, int L, int M) {
    stats_kernel<<<Nn[L], 256, 0, stream>>>(Y, M, Nn[L], gg[L], bb[L], acoef, ccoef);
  };

  // ---- Set abstraction 1 -------------------------------------------------
  fps_kernel<<<B, 256, 0, stream>>>(xyz, fidx, N, S);
  gather_nxyz_kernel<<<cdiv((long long)BS * 3, 256), 256, 0, stream>>>(
      xyz, fidx, nxyz, B, N, S);
  ballq_kernel<<<cdiv((long long)BS, 256), 256, 0, stream>>>(
      xyz, nxyz, gidx, B, N, S, K, radius * radius);
  group_pack_kernel<<<cdiv((long long)M1, 256), 256, 0, stream>>>(
      xyz, nxyz, gidx, XhA, B, N, S, K);

  // L0: 3->64
  gemm(XhA, 0, Ybuf, M1); bnstats(Ybuf, 0, M1);
  normrelu_pack_kernel<<<cdiv((long long)M1 * 64, 256), 256, 0, stream>>>(
      Ybuf, acoef, ccoef, XhB, M1, 64, 64);
  // L1: 64->64
  gemm(XhB, 1, Ybuf, M1); bnstats(Ybuf, 1, M1);
  normrelu_pack_kernel<<<cdiv((long long)M1 * 64, 256), 256, 0, stream>>>(
      Ybuf, acoef, ccoef, XhA, M1, 64, 64);
  // L2: 64->128, then BN+ReLU+maxpool over K=32
  gemm(XhA, 2, Ybuf, M1); bnstats(Ybuf, 2, M1);
  normrelu_maxpool_kernel<<<cdiv((long long)BS * 128, 256), 256, 0, stream>>>(
      Ybuf, acoef, ccoef, l1pts, BS, K, 128);

  // ---- Set abstraction 2 (group_all) ------------------------------------
  concat_pack_kernel<<<cdiv((long long)BS * 160, 256), 256, 0, stream>>>(
      nxyz, l1pts, XhA, BS, 128, 160);

  // L3: 131->128
  gemm(XhA, 3, Ybuf, BS); bnstats(Ybuf, 3, BS);
  normrelu_pack_kernel<<<cdiv((long long)BS * 128, 256), 256, 0, stream>>>(
      Ybuf, acoef, ccoef, XhB, BS, 128, 128);
  // L4: 128->128
  gemm(XhB, 4, Ybuf, BS); bnstats(Ybuf, 4, BS);
  normrelu_pack_kernel<<<cdiv((long long)BS * 128, 256), 256, 0, stream>>>(
      Ybuf, acoef, ccoef, XhA, BS, 128, 128);
  // L5: 128->256, then BN+ReLU+maxpool over S=512 -> d_out (32x256)
  gemm(XhA, 5, Ybuf, BS); bnstats(Ybuf, 5, BS);
  normrelu_maxpool_kernel<<<cdiv((long long)B * 256, 256), 256, 0, stream>>>(
      Ybuf, acoef, ccoef, (float*)d_out, B, S, 256);
}
